// MetricalGNN_56049323213470
// MI455X (gfx1250) — matrix operand order repaired and checked by reading, not compiled, not run on hardware
//
#include <hip/hip_runtime.h>
#include <hip/hip_bf16.h>

// ---------------------------------------------------------------------------
// Types for CDNA5 WMMA (wave32, v_wmma_f32_16x16x32_bf16)
// ---------------------------------------------------------------------------
typedef __bf16 bf16_t;
typedef __attribute__((ext_vector_type(16))) __bf16 v16bf;
typedef __attribute__((ext_vector_type(8)))  __bf16 bf16x8;
typedef __attribute__((ext_vector_type(8)))  float  v8f;

union WmmaRegs { v16bf v; bf16x8 h[2]; };

// ---------------------------------------------------------------------------
// Generic strided-batched GEMM:  C[M,N](f32) = A[M,K](bf16) * B[N,K](bf16)^T
// Optional: +bias[N], +=C (accum), ReLU.  Per-batch offsets for A/B/C are
// (b / div) * strideHi + (b % div) * strideLo  (element units) so the 2048
// attention sub-GEMMs can address q/k/v/o slices of the packed QKV buffer.
//
// NT (number of 16-wide N tiles per wave) is a compile-time parameter so the
// k-loop is branch-free.  Per k-step we issue ALL loads first (A fragment +
// NT B fragments into distinct registers) and only then the NT WMMAs, so the
// scheduler can clause the b128 loads and retire them with partial
// s_wait_loadcnt values while earlier WMMAs execute.  The A fragment is
// reused by all NT v_wmma issues (4x A-traffic reduction at NT=4).
// Block = 128 threads = 4 waves (4 M tiles); grid.x covers N/(16*NT) exactly.
// The only EXEC-affecting branch is the wave-uniform M guard.
// ---------------------------------------------------------------------------
struct GemmP {
  const bf16_t* A; long lda; long aDiv; long aSH; long aSL;
  const bf16_t* B; long ldb; long bDiv; long bSH; long bSL;
  float*        C; long ldc; long cDiv; long cSH; long cSL;
  const float* bias;
  int M, N, K;
  int accum, relu;
};

template <int NT>
__global__ __launch_bounds__(128) void gemm_bf16_nt(GemmP p) {
  const int wave = threadIdx.x >> 5;
  const int lane = threadIdx.x & 31;
  const int mt = blockIdx.y * 4 + wave;
  if (mt * 16 >= p.M) return;            // uniform per-wave: EXEC stays full
  const int nt0 = blockIdx.x * NT;       // base N tile of this wave's strip
  const long b  = blockIdx.z;
  const int row = lane & 15;             // A row / B col within tile
  const int hi  = lane >> 4;

  const bf16_t* Ap = p.A + (b / p.aDiv) * p.aSH + (b % p.aDiv) * p.aSL
                         + (long)(mt * 16 + row) * p.lda;
  const bf16_t* Bp = p.B + (b / p.bDiv) * p.bSH + (b % p.bDiv) * p.bSL
                         + (long)(nt0 * 16 + row) * p.ldb;
  const long bTile = 16 * p.ldb;         // B advance per N tile

  v8f acc[NT];
#pragma unroll
  for (int j = 0; j < NT; ++j) acc[j] = (v8f){};

  for (int k0 = 0; k0 < p.K; k0 += 32) {
    // prefetch next K tiles toward the caches (global_prefetch_b8)
    if (k0 + 128 < p.K) {
      __builtin_prefetch(Ap + k0 + 128, 0, 3);
      __builtin_prefetch(Bp + k0 + 128, 0, 3);
    }
    // ---- issue ALL loads first (one clause), then the WMMAs ----
    WmmaRegs a, bm[NT];
    // A 16x32 bf16: lanes0-15 K={0..7,16..23}, lanes16-31 K={8..15,24..31}
    a.h[0] = *(const bf16x8*)(Ap + k0 + hi * 8);
    a.h[1] = *(const bf16x8*)(Ap + k0 + 16 + hi * 8);
#pragma unroll
    for (int j = 0; j < NT; ++j) {
      // B 32x16 bf16: lane = column, lanes0-15 K={0..7,16..23}, lanes16-31 K={8..15,24..31}
      bm[j].h[0] = *(const bf16x8*)(Bp + j * bTile + k0 + hi * 8);
      bm[j].h[1] = *(const bf16x8*)(Bp + j * bTile + k0 + 16 + hi * 8);
    }
#pragma unroll
    for (int j = 0; j < NT; ++j)
      acc[j] = __builtin_amdgcn_wmma_f32_16x16x32_bf16(false, a.v, false, bm[j].v,
                                                       (short)0, acc[j], false, false);
  }

  float* Cb = p.C + (b / p.cDiv) * p.cSH + (b % p.cDiv) * p.cSL;
  // C/D 16x16 f32: VGPR j -> M = hi*8 + j, N = lane&15
#pragma unroll
  for (int jt = 0; jt < NT; ++jt) {
    const int c = (nt0 + jt) * 16 + row;
    for (int j = 0; j < 8; ++j) {
      const int r = mt * 16 + hi * 8 + j;
      if (r < p.M && c < p.N) {
        float v = acc[jt][j];
        if (p.bias)  v += p.bias[c];
        float* cp = Cb + (long)r * p.ldc + c;
        if (p.accum) v += *cp;
        if (p.relu)  v = fmaxf(v, 0.f);
        *cp = v;
      }
    }
  }
}

// ---------------------------------------------------------------------------
// Elementwise / reduction helper kernels
// ---------------------------------------------------------------------------
__global__ __launch_bounds__(256) void zero_f32(float* x, long n) {
  long i = (long)blockIdx.x * blockDim.x + threadIdx.x;
  if (i < n) x[i] = 0.f;
}

__global__ __launch_bounds__(256) void f32_to_bf16_k(bf16_t* d, const float* s, long n) {
  long i = (long)blockIdx.x * blockDim.x + threadIdx.x;
  if (i < n) d[i] = (bf16_t)s[i];
}

__global__ __launch_bounds__(256) void seg_add(float* agg, float* cnt, const float* x,
                                               const int* src, const int* dst,
                                               long E, int Din) {
  long i = (long)blockIdx.x * blockDim.x + threadIdx.x;
  if (i >= E * (long)Din) return;
  long e = i / Din; int d = (int)(i - e * Din);
  int s = src[e], t = dst[e];
  atomicAdd(&agg[(long)t * Din + d], x[(long)s * Din + d]);
  if (d == 0) atomicAdd(&cnt[t], 1.0f);
}

__global__ __launch_bounds__(256) void seg_div(float* agg, const float* cnt,
                                               long N, int Din) {
  long i = (long)blockIdx.x * blockDim.x + threadIdx.x;
  if (i >= N * (long)Din) return;
  long r = i / Din;
  agg[i] /= fmaxf(cnt[r], 1.0f);
}

__global__ __launch_bounds__(256) void scale_relu(float* x, long n, float s, int relu) {
  long i = (long)blockIdx.x * blockDim.x + threadIdx.x;
  if (i >= n) return;
  float v = x[i] * s;
  if (relu) v = fmaxf(v, 0.f);
  x[i] = v;
}

// LayerNorm over D=256; one wave per row (wave32: 8 elements per lane).
__global__ __launch_bounds__(256) void layernorm256(const float* x, const float* res,
                                                    const float* g, const float* bb,
                                                    float* y, int rows) {
  const int wave = threadIdx.x >> 5, lane = threadIdx.x & 31;
  const int row = blockIdx.x * 8 + wave;
  if (row >= rows) return;
  const long base = (long)row * 256;
  float v[8]; float s = 0.f;
  for (int i = 0; i < 8; ++i) {
    float t = x[base + lane + i * 32];
    if (res) t += res[base + lane + i * 32];
    v[i] = t; s += t;
  }
  for (int o = 16; o > 0; o >>= 1) s += __shfl_xor(s, o, 32);
  const float mean = s * (1.f / 256.f);
  float q = 0.f;
  for (int i = 0; i < 8; ++i) { float d = v[i] - mean; q += d * d; }
  for (int o = 16; o > 0; o >>= 1) q += __shfl_xor(q, o, 32);
  const float inv = rsqrtf(q * (1.f / 256.f) + 1e-5f);
  for (int i = 0; i < 8; ++i)
    y[base + lane + i * 32] = (v[i] - mean) * inv * g[lane + i * 32] + bb[lane + i * 32];
}

// softmax over rows of length 64 (attention over the 64-graph sequence axis),
// scale folded in; writes bf16 directly (feeds the A*V WMMA GEMM).
__global__ __launch_bounds__(256) void softmax64_bf16(const float* x, bf16_t* y,
                                                      long rows, float scale) {
  const int wave = threadIdx.x >> 5, lane = threadIdx.x & 31;
  const long row = (long)blockIdx.x * 8 + wave;
  if (row >= rows) return;
  const long base = row * 64;
  float a = x[base + lane] * scale;
  float b = x[base + lane + 32] * scale;
  float m = fmaxf(a, b);
  for (int o = 16; o > 0; o >>= 1) m = fmaxf(m, __shfl_xor(m, o, 32));
  float ea = __expf(a - m), eb = __expf(b - m);
  float s = ea + eb;
  for (int o = 16; o > 0; o >>= 1) s += __shfl_xor(s, o, 32);
  const float inv = 1.f / s;
  y[base + lane]      = (bf16_t)(ea * inv);
  y[base + lane + 32] = (bf16_t)(eb * inv);
}

// v_t[((n_b*8+h)*32 + hd)*64 + t] = qkv[(t*256 + n_b)*768 + 512 + h*32 + hd]
__global__ __launch_bounds__(256) void build_vt(const float* qkv, bf16_t* vt, long total) {
  long i = (long)blockIdx.x * blockDim.x + threadIdx.x;
  if (i >= total) return;
  int t  = (int)(i & 63);
  int hd = (int)((i >> 6) & 31);
  int h  = (int)((i >> 11) & 7);
  long nb = i >> 14;
  vt[i] = (bf16_t)qkv[(((long)t * 256 + nb) * 768) + 512 + h * 32 + hd];
}

// new_note[dst[e]] = x_note2[dst[e]] + h[src[e]]  (dst covers notes uniquely)
__global__ __launch_bounds__(256) void note_scatter(const float* xn, const float* hb,
                                                    const int* src, const int* dst,
                                                    float* out, long E) {
  long i = (long)blockIdx.x * blockDim.x + threadIdx.x;
  if (i >= E * 256) return;
  long e = i >> 8; int d = (int)(i & 255);
  int sb = src[e], dn = dst[e];
  out[(long)dn * 256 + d] = xn[(long)dn * 256 + d] + hb[(long)sb * 256 + d];
}

// mlp input: new_note.reshape(64,1024,256)[:, :512] -> bf16 [32768,256]
__global__ __launch_bounds__(256) void gather_half_bf16(const float* src, bf16_t* dst,
                                                        long n) {
  long i = (long)blockIdx.x * blockDim.x + threadIdx.x;
  if (i >= n) return;
  int d = (int)(i & 255);
  long r = i >> 8;
  long g = r >> 9, l = r & 511;
  dst[i] = (bf16_t)src[((g << 10) + l) * 256 + d];
}

// ---------------------------------------------------------------------------
// Host orchestration
// ---------------------------------------------------------------------------
static inline long cdivl(long a, long b) { return (a + b - 1) / b; }

extern "C" void kernel_launch(void* const* d_in, const int* in_sizes, int n_in,
                              void* d_out, int out_size, void* d_ws, size_t ws_size,
                              hipStream_t stream) {
  (void)n_in; (void)out_size; (void)ws_size;
  const int DIN = 64, D = 256, FF = 2048, G = 64, NPG = 1024, LPG = 256;
  const long NOTES = (long)G * NPG;      // 65536
  const long BEATS = (long)G * LPG;      // 16384
  const long TOK   = BEATS;              // 16384 transformer tokens
  const long AB    = (long)LPG * 8;      // 2048 attention batches (n_b, head)
  const long E_NN = in_sizes[2] / 2, E_NB = in_sizes[3] / 2;
  const long E_BN = in_sizes[4] / 2, E_BB = in_sizes[5] / 2;

  const float* x_note = (const float*)d_in[0];
  const float* x_beat = (const float*)d_in[1];
  const int* e_nn_s = (const int*)d_in[2]; const int* e_nn_d = e_nn_s + E_NN;
  const int* e_nb_s = (const int*)d_in[3]; const int* e_nb_d = e_nb_s + E_NB;
  const int* e_bn_s = (const int*)d_in[4]; const int* e_bn_d = e_bn_s + E_BN;
  const int* e_bb_s = (const int*)d_in[5]; const int* e_bb_d = e_bb_s + E_BB;

  // params flattened in setup_inputs() insertion order starting at d_in[9]:
  //  gnn L0: nn(Wl,bl,Wr) bn(..) nb(..) bb(..) ln_note_g/b ln_beat_g/b  -> 9..24
  //  gnn L1: nn bn nb bb                                                 -> 25..36
  //  tr t=0..2 (base 37+12t): Win,bin,Wout,bout,W1,b1,W2,b2,ln1g,ln1b,ln2g,ln2b
  //  73: tr_norm_g  74: tr_norm_b
  //  mlp: 75 W1, 76 b1, 77 lng, 78 lnb, 79 W2, 80 b2
  auto P = [&](int i) { return (const float*)d_in[i]; };

  // ---- bump allocator over d_ws ----
  char* wsbase = (char*)d_ws;
  size_t cur = 0;
  auto alloc = [&](size_t bytes) {
    cur = (cur + 255) & ~(size_t)255;
    void* p = wsbase + cur;
    cur += bytes;
    return p;
  };
  float*  agg_nn_f = (float*)alloc(NOTES * D * 4);
  float*  agg_bn_f = (float*)alloc(NOTES * D * 4);
  float*  agg_nb_f = (float*)alloc(BEATS * D * 4);
  float*  agg_bb_f = (float*)alloc(BEATS * D * 4);
  float*  cnt_n    = (float*)alloc(NOTES * 4);
  float*  cnt_b    = (float*)alloc(BEATS * 4);
  float*  noteA    = (float*)alloc(NOTES * D * 4);
  float*  noteB    = (float*)alloc(NOTES * D * 4);
  float*  beatA    = (float*)alloc(BEATS * D * 4);
  float*  beatB    = (float*)alloc(BEATS * D * 4);
  bf16_t* xn_bf    = (bf16_t*)alloc(NOTES * D * 2);
  bf16_t* xb_bf    = (bf16_t*)alloc(BEATS * D * 2);
  bf16_t* agg_nn_bf = (bf16_t*)alloc(NOTES * D * 2);
  bf16_t* agg_bn_bf = (bf16_t*)alloc(NOTES * D * 2);
  bf16_t* agg_nb_bf = (bf16_t*)alloc(BEATS * D * 2);
  bf16_t* agg_bb_bf = (bf16_t*)alloc(BEATS * D * 2);
  bf16_t* wbuf     = (bf16_t*)alloc((size_t)FF * D * 2);   // largest weight
  bf16_t* h_bf     = (bf16_t*)alloc(TOK * D * 2);
  float*  qkv      = (float*)alloc(TOK * 3 * D * 4);
  bf16_t* qkv_bf   = (bf16_t*)alloc(TOK * 3 * D * 2);
  float*  scores   = (float*)alloc(AB * 64 * 64 * 4);
  bf16_t* a_bf     = (bf16_t*)alloc(AB * 64 * 64 * 2);
  bf16_t* v_t      = (bf16_t*)alloc(AB * 32 * 64 * 2);
  float*  attn_c   = (float*)alloc(TOK * D * 4);
  bf16_t* o_bf     = (bf16_t*)alloc(TOK * D * 2);
  float*  attn_out = (float*)alloc(TOK * D * 4);
  float*  hmid     = (float*)alloc(TOK * D * 4);
  float*  tr0      = (float*)alloc(TOK * D * 4);
  float*  tr1      = (float*)alloc(TOK * D * 4);
  float*  hfin     = (float*)alloc(TOK * D * 4);
  float*  ff1      = (float*)alloc(TOK * FF * 4);
  bf16_t* ff1_bf   = (bf16_t*)alloc(TOK * FF * 2);
  float*  new_note = (float*)alloc(NOTES * D * 4);
  bf16_t* mlp_in_bf = (bf16_t*)alloc((long)32768 * D * 2);
  float*  mlp1     = (float*)alloc((long)32768 * D * 4);
  bf16_t* mlp1_bf  = (bf16_t*)alloc((long)32768 * D * 2);

  // ---- launch helpers ----
  auto zero = [&](float* p, long n) {
    zero_f32<<<dim3((unsigned)cdivl(n, 256)), 256, 0, stream>>>(p, n);
  };
  auto cvt = [&](const float* s, bf16_t* d, long n) {
    f32_to_bf16_k<<<dim3((unsigned)cdivl(n, 256)), 256, 0, stream>>>(d, s, n);
  };
  auto wB = [&](const float* w, long n) {  // weight -> bf16 staging
    cvt(w, wbuf, n);
    return (const bf16_t*)wbuf;
  };
  auto gemm = [&](const bf16_t* A, long lda, const bf16_t* B, long ldb,
                  float* C, long ldc, int M, int N, int K, const float* bias,
                  bool accum, bool relu, int batch = 1,
                  long aDiv = 1, long aSH = 0, long aSL = 0,
                  long bDiv = 1, long bSH = 0, long bSL = 0,
                  long cDiv = 1, long cSH = 0, long cSL = 0) {
    GemmP p;
    p.A = A; p.lda = lda; p.aDiv = aDiv; p.aSH = aSH; p.aSL = aSL;
    p.B = B; p.ldb = ldb; p.bDiv = bDiv; p.bSH = bSH; p.bSL = bSL;
    p.C = C; p.ldc = ldc; p.cDiv = cDiv; p.cSH = cSH; p.cSL = cSL;
    p.bias = bias; p.M = M; p.N = N; p.K = K;
    p.accum = accum ? 1 : 0; p.relu = relu ? 1 : 0;
    const int ntiles = N >> 4;                 // all N are multiples of 16
    int nt = (ntiles % 4 == 0) ? 4
           : (ntiles % 3 == 0) ? 3
           : (ntiles % 2 == 0) ? 2 : 1;
    dim3 grid((unsigned)(ntiles / nt), (unsigned)((M + 63) / 64), (unsigned)batch);
    switch (nt) {
      case 4: gemm_bf16_nt<4><<<grid, 128, 0, stream>>>(p); break;
      case 3: gemm_bf16_nt<3><<<grid, 128, 0, stream>>>(p); break;
      case 2: gemm_bf16_nt<2><<<grid, 128, 0, stream>>>(p); break;
      default: gemm_bf16_nt<1><<<grid, 128, 0, stream>>>(p); break;
    }
  };
  auto ln = [&](const float* x, const float* res, const float* g, const float* b,
                float* y, long rows) {
    layernorm256<<<dim3((unsigned)cdivl(rows, 8)), 256, 0, stream>>>(x, res, g, b, y, (int)rows);
  };

  // ---- hetero SAGE layer ----
  auto sage_layer = [&](int din, const float* xn_f, const float* xb_f,
                        float* note_out, float* beat_out, int pb, bool act,
                        int lnb /* ln base index (only if act) */) {
    // note aggregations
    zero(agg_nn_f, NOTES * din); zero(cnt_n, NOTES);
    seg_add<<<dim3((unsigned)cdivl(E_NN * din, 256)), 256, 0, stream>>>(
        agg_nn_f, cnt_n, xn_f, e_nn_s, e_nn_d, E_NN, din);
    seg_div<<<dim3((unsigned)cdivl(NOTES * din, 256)), 256, 0, stream>>>(
        agg_nn_f, cnt_n, NOTES, din);
    zero(agg_bn_f, NOTES * din); zero(cnt_n, NOTES);
    seg_add<<<dim3((unsigned)cdivl(E_BN * din, 256)), 256, 0, stream>>>(
        agg_bn_f, cnt_n, xb_f, e_bn_s, e_bn_d, E_BN, din);
    seg_div<<<dim3((unsigned)cdivl(NOTES * din, 256)), 256, 0, stream>>>(
        agg_bn_f, cnt_n, NOTES, din);
    // beat aggregations
    zero(agg_nb_f, BEATS * din); zero(cnt_b, BEATS);
    seg_add<<<dim3((unsigned)cdivl(E_NB * din, 256)), 256, 0, stream>>>(
        agg_nb_f, cnt_b, xn_f, e_nb_s, e_nb_d, E_NB, din);
    seg_div<<<dim3((unsigned)cdivl(BEATS * din, 256)), 256, 0, stream>>>(
        agg_nb_f, cnt_b, BEATS, din);
    zero(agg_bb_f, BEATS * din); zero(cnt_b, BEATS);
    seg_add<<<dim3((unsigned)cdivl(E_BB * din, 256)), 256, 0, stream>>>(
        agg_bb_f, cnt_b, xb_f, e_bb_s, e_bb_d, E_BB, din);
    seg_div<<<dim3((unsigned)cdivl(BEATS * din, 256)), 256, 0, stream>>>(
        agg_bb_f, cnt_b, BEATS, din);

    cvt(agg_nn_f, agg_nn_bf, NOTES * din);
    cvt(agg_bn_f, agg_bn_bf, NOTES * din);
    cvt(agg_nb_f, agg_nb_bf, BEATS * din);
    cvt(agg_bb_f, agg_bb_bf, BEATS * din);
    cvt(xn_f, xn_bf, NOTES * din);
    cvt(xb_f, xb_bf, BEATS * din);

    // note = 0.5*(agg_nn@Wl_nn^T + bl_nn + x_n@Wr_nn^T + agg_bn@Wl_bn^T + bl_bn + x_n@Wr_bn^T)
    gemm(agg_nn_bf, din, wB(P(pb + 0), (long)D * din), din, note_out, D,
         (int)NOTES, D, din, P(pb + 1), false, false);
    gemm(xn_bf, din, wB(P(pb + 2), (long)D * din), din, note_out, D,
         (int)NOTES, D, din, nullptr, true, false);
    gemm(agg_bn_bf, din, wB(P(pb + 3), (long)D * din), din, note_out, D,
         (int)NOTES, D, din, P(pb + 4), true, false);
    gemm(xn_bf, din, wB(P(pb + 5), (long)D * din), din, note_out, D,
         (int)NOTES, D, din, nullptr, true, false);
    scale_relu<<<dim3((unsigned)cdivl(NOTES * D, 256)), 256, 0, stream>>>(
        note_out, NOTES * D, 0.5f, act ? 1 : 0);
    if (act) ln(note_out, nullptr, P(lnb + 0), P(lnb + 1), note_out, NOTES);

    // beat = 0.5*(agg_nb@Wl_nb^T + bl_nb + x_b@Wr_nb^T + agg_bb@Wl_bb^T + bl_bb + x_b@Wr_bb^T)
    gemm(agg_nb_bf, din, wB(P(pb + 6), (long)D * din), din, beat_out, D,
         (int)BEATS, D, din, P(pb + 7), false, false);
    gemm(xb_bf, din, wB(P(pb + 8), (long)D * din), din, beat_out, D,
         (int)BEATS, D, din, nullptr, true, false);
    gemm(agg_bb_bf, din, wB(P(pb + 9), (long)D * din), din, beat_out, D,
         (int)BEATS, D, din, P(pb + 10), true, false);
    gemm(xb_bf, din, wB(P(pb + 11), (long)D * din), din, beat_out, D,
         (int)BEATS, D, din, nullptr, true, false);
    scale_relu<<<dim3((unsigned)cdivl(BEATS * D, 256)), 256, 0, stream>>>(
        beat_out, BEATS * D, 0.5f, act ? 1 : 0);
    if (act) ln(beat_out, nullptr, P(lnb + 2), P(lnb + 3), beat_out, BEATS);
  };

  // ---- GNN ----
  sage_layer(DIN, x_note, x_beat, noteA, beatA, 9, true, 21);
  sage_layer(D,   noteA,  beatA,  noteB, beatB, 25, false, 0);

  // ---- Transformer (3 post-norm encoder layers; S=64 graphs, batch=256) ----
  const float* hin = beatB;
  float* hout = tr0;
  const float inv_sqrt_hd = 0.17677669529663687f;  // 1/sqrt(32)
  for (int t = 0; t < 3; ++t) {
    const int pb = 37 + t * 12;
    hout = (t & 1) ? tr1 : tr0;
    cvt(hin, h_bf, TOK * D);
    // QKV projection  [16384,768]
    gemm(h_bf, D, wB(P(pb + 0), (long)3 * D * D), D, qkv, 3 * D,
         (int)TOK, 3 * D, D, P(pb + 1), false, false);
    cvt(qkv, qkv_bf, TOK * 3 * D);
    // scores[b=(n_b*8+h)][s][t] = q . k  (2048 batched 64x64x32 WMMA GEMMs)
    gemm(qkv_bf, (long)LPG * 3 * D, qkv_bf + D, (long)LPG * 3 * D,
         scores, 64, 64, 64, 32, nullptr, false, false, (int)AB,
         /*A*/ 8, 3 * D, 32, /*B*/ 8, 3 * D, 32, /*C*/ 1, 64 * 64, 0);
    softmax64_bf16<<<dim3((unsigned)cdivl(AB * 64, 8)), 256, 0, stream>>>(
        scores, a_bf, AB * 64, inv_sqrt_hd);
    build_vt<<<dim3((unsigned)cdivl(AB * 32 * 64, 256)), 256, 0, stream>>>(
        qkv, v_t, AB * 32 * 64);
    // o[s,hd] = sum_t a[s,t] v[t,hd]  -> concat heads into attn_c [16384,256]
    gemm(a_bf, 64, v_t, 64, attn_c, (long)LPG * D, 64, 32, 64, nullptr,
         false, false, (int)AB,
         /*A*/ 1, 64 * 64, 0, /*B*/ 1, 32 * 64, 0, /*C*/ 8, D, 32);
    cvt(attn_c, o_bf, TOK * D);
    // output projection
    gemm(o_bf, D, wB(P(pb + 2), (long)D * D), D, attn_out, D,
         (int)TOK, D, D, P(pb + 3), false, false);
    // hmid = LN(hin + attn)
    ln(attn_out, hin, P(pb + 8), P(pb + 9), hmid, TOK);
    // FF
    cvt(hmid, h_bf, TOK * D);
    gemm(h_bf, D, wB(P(pb + 4), (long)FF * D), D, ff1, FF,
         (int)TOK, FF, D, P(pb + 5), false, true);        // ReLU fused
    cvt(ff1, ff1_bf, TOK * FF);
    gemm(ff1_bf, FF, wB(P(pb + 6), (long)D * FF), FF, attn_out, D,
         (int)TOK, D, FF, P(pb + 7), false, false);
    // hout = LN(hmid + ff2)
    ln(attn_out, hmid, P(pb + 10), P(pb + 11), hout, TOK);
    hin = hout;
  }
  // final transformer norm
  ln(hin, nullptr, P(73), P(74), hfin, TOK);

  // ---- scatter beats back to notes; residual with GNN note output ----
  hipMemcpyAsync(new_note, noteB, NOTES * D * 4, hipMemcpyDeviceToDevice, stream);
  note_scatter<<<dim3((unsigned)cdivl(E_BN * 256, 256)), 256, 0, stream>>>(
      noteB, hfin, e_bn_s, e_bn_d, new_note, E_BN);

  // ---- head MLP on first 512 notes of each graph (32768 rows) ----
  const long MR = (long)G * 512;
  gather_half_bf16<<<dim3((unsigned)cdivl(MR * D, 256)), 256, 0, stream>>>(
      new_note, mlp_in_bf, MR * D);
  gemm(mlp_in_bf, D, wB(P(75), (long)D * D), D, mlp1, D,
       (int)MR, D, D, P(76), false, true);                // ReLU fused
  ln(mlp1, nullptr, P(77), P(78), mlp1, MR);
  cvt(mlp1, mlp1_bf, MR * D);
  gemm(mlp1_bf, D, wB(P(79), (long)D * D), D, (float*)d_out, D,
       (int)MR, D, D, P(80), false, false);
}
